// MoEFFNLM_49460843381516
// MI455X (gfx1250) — matrix operand order repaired and checked by reading, not compiled, not run on hardware
//
#include <hip/hip_runtime.h>
#include <hip/hip_bf16.h>
#include <math.h>

// Model config (matches reference)
#define VV 32000
#define CC 768
#define HH 12
#define LL 2
#define EE 4
#define BB 2
#define TT 1024
#define HD 64
#define FF 3072
#define NTOK (BB * TT)   // 2048 tokens

typedef __attribute__((ext_vector_type(16))) _Float16 v16h;
typedef __attribute__((ext_vector_type(8)))  _Float16 v8h;
typedef __attribute__((ext_vector_type(8)))  float    v8f;
typedef __attribute__((ext_vector_type(8)))  unsigned int v8u;

union Frag { v8u u; v16h h; };

// ---------------------------------------------------------------------------
// Embedding: x[n,c] = tok_emb[idx[n],c] + pos_emb[n % T, c]   (float4 wide)
// ---------------------------------------------------------------------------
__global__ void k_embed(const int* __restrict__ idx, const float* __restrict__ tok,
                        const float* __restrict__ pos, float* __restrict__ x) {
  int q = blockIdx.x * blockDim.x + threadIdx.x;      // quad index
  if (q >= NTOK * CC / 4) return;
  int n = q / (CC / 4), c4 = (q - n * (CC / 4)) * 4;
  int t = n & (TT - 1);
  const float4 tv = *(const float4*)&tok[(size_t)idx[n] * CC + c4];
  const float4 pv = *(const float4*)&pos[(size_t)t * CC + c4];
  float4 r; r.x = tv.x + pv.x; r.y = tv.y + pv.y; r.z = tv.z + pv.z; r.w = tv.w + pv.w;
  *(float4*)&x[(size_t)n * CC + c4] = r;
}

// ---------------------------------------------------------------------------
// LayerNorm over C=768, one block (256 thr) per row.
// ---------------------------------------------------------------------------
__global__ void k_ln(const float* __restrict__ x, const float* __restrict__ s,
                     const float* __restrict__ b, float* __restrict__ y) {
  __shared__ float red[256];
  int n = blockIdx.x;
  const float* xr = x + (size_t)n * CC;
  float lsum = 0.f;
  for (int c = threadIdx.x; c < CC; c += 256) lsum += xr[c];
  red[threadIdx.x] = lsum; __syncthreads();
  for (int off = 128; off > 0; off >>= 1) {
    if ((int)threadIdx.x < off) red[threadIdx.x] += red[threadIdx.x + off];
    __syncthreads();
  }
  float mean = red[0] * (1.0f / CC);
  __syncthreads();
  float lss = 0.f;
  for (int c = threadIdx.x; c < CC; c += 256) { float d = xr[c] - mean; lss += d * d; }
  red[threadIdx.x] = lss; __syncthreads();
  for (int off = 128; off > 0; off >>= 1) {
    if ((int)threadIdx.x < off) red[threadIdx.x] += red[threadIdx.x + off];
    __syncthreads();
  }
  float inv = rsqrtf(red[0] * (1.0f / CC) + 1e-5f);
  for (int c = threadIdx.x; c < CC; c += 256)
    y[(size_t)n * CC + c] = (xr[c] - mean) * inv * s[c] + b[c];
}

// ---------------------------------------------------------------------------
// WMMA GEMM: C[M,N] (+)= epilogue( A[M,K] @ B[K,N] [+ bias] ), M = NTOK.
// N, K are template constants so every global load folds its stride into the
// 24-bit instruction offset (one base address, clause of immediate-offset
// loads, no xcnt add-chains).  Block tile 64x64, 128 threads = 4 waves, each
// wave a 32x32 macro-tile as 2x2 v_wmma_f32_16x16x32_f16 (f32 accum).
// Global loads for step k+32 are issued before the WMMA of step k.
// NTB: non-temporal B loads + non-temporal C stores (streaming logits GEMM).
// ---------------------------------------------------------------------------
#define BM 64
#define BN 64
#define BK 32
#define LDP (BK + 8)     // padded row stride (80B, 16B-aligned)

template <int N, int K, bool BIAS, bool RELU, bool ACC, bool GATE, bool NTB>
__global__ __launch_bounds__(128)
void k_gemm(const float* __restrict__ A, const float* __restrict__ B,
            const float* __restrict__ bias, float* __restrict__ C,
            const int* __restrict__ gate, int esel) {
  __shared__ _Float16 As[BM][LDP];
  __shared__ _Float16 Bs[BN][LDP];    // B^T tile: [n][k]

  const int tid  = threadIdx.x;
  const int lane = tid & 31;
  const int wid  = tid >> 5;
  const int lo   = lane & 15;
  const int hi   = lane >> 4;
  const int m0 = blockIdx.y * BM, n0 = blockIdx.x * BN;
  const int wr = (wid >> 1) * 32, wc = (wid & 1) * 32;

  // Segment decomposition (2 segments x 128 threads = 256 segs of 8 halves)
  // A: seg -> row am (64), 8 consecutive k at akq; float4-pair global loads.
  // B: seg -> col bn (64), 8 consecutive k at bkq; 8 coalesced b32 loads with
  //    compile-time row offsets j*N.
  v8h aReg[2], bReg[2];

  auto load_tiles = [&](int k0) {
#pragma unroll
    for (int i = 0; i < 2; ++i) {
      int seg = i * 128 + tid;
      int am = seg >> 2, akq = (seg & 3) * 8;
      const float* ap = &A[(size_t)(m0 + am) * K + k0 + akq];
      float4 f0 = *(const float4*)ap;
      float4 f1 = *(const float4*)(ap + 4);
      v8h h;
      h[0] = (_Float16)f0.x; h[1] = (_Float16)f0.y;
      h[2] = (_Float16)f0.z; h[3] = (_Float16)f0.w;
      h[4] = (_Float16)f1.x; h[5] = (_Float16)f1.y;
      h[6] = (_Float16)f1.z; h[7] = (_Float16)f1.w;
      aReg[i] = h;
      int bn = seg & 63, bkq = (seg >> 6) * 8;
      const float* bp = &B[(size_t)(k0 + bkq) * N + n0 + bn];
      v8h g;
#pragma unroll
      for (int j = 0; j < 8; ++j) {
        float bv = NTB ? __builtin_nontemporal_load(bp + (size_t)j * N)
                       : bp[(size_t)j * N];
        g[j] = (_Float16)bv;
      }
      bReg[i] = g;
    }
  };
  auto store_tiles = [&]() {
#pragma unroll
    for (int i = 0; i < 2; ++i) {
      int seg = i * 128 + tid;
      *(v8h*)&As[seg >> 2][(seg & 3) * 8] = aReg[i];
      *(v8h*)&Bs[seg & 63][(seg >> 6) * 8] = bReg[i];
    }
  };

  v8f acc[2][2] = {};
  load_tiles(0);
  for (int k0 = 0; k0 < K; k0 += BK) {
    store_tiles();
    __syncthreads();
    if (k0 + BK < K) load_tiles(k0 + BK);   // overlap next global loads w/ WMMA

    // Fragments per ISA 7.12.2: A row-of-K per lane, B column-of-K per lane;
    // lanes 16-31 take k 8-15 and 24-31.
    Frag af[2], bf[2];
#pragma unroll
    for (int mi = 0; mi < 2; ++mi)
#pragma unroll
      for (int v = 0; v < 8; ++v) {
        int kk = ((v & 3) * 2) + ((v >> 2) * 16) + hi * 8;   // even -> u32 pair
        af[mi].u[v] = *(const unsigned int*)&As[wr + mi * 16 + lo][kk];
      }
#pragma unroll
    for (int ni = 0; ni < 2; ++ni)
#pragma unroll
      for (int v = 0; v < 8; ++v)
        bf[ni].u[v] = *(const unsigned int*)&Bs[wc + ni * 16 + lo][((v & 3) * 2) + ((v >> 2) * 16) + hi * 8];

#pragma unroll
    for (int mi = 0; mi < 2; ++mi)
#pragma unroll
      for (int ni = 0; ni < 2; ++ni)
        acc[mi][ni] = __builtin_amdgcn_wmma_f32_16x16x32_f16(
            false, af[mi].h, false, bf[ni].h, (short)0, acc[mi][ni], false, false);
    __syncthreads();
  }

  // Epilogue: C/D layout — lane lo = column, VGPR r -> row r + hi*8
#pragma unroll
  for (int mi = 0; mi < 2; ++mi)
#pragma unroll
    for (int ni = 0; ni < 2; ++ni) {
      int mbase = m0 + wr + mi * 16 + hi * 8;
      int n = n0 + wc + ni * 16 + lo;
      float bs = BIAS ? bias[n] : 0.f;
#pragma unroll
      for (int r = 0; r < 8; ++r) {
        int mr = mbase + r;
        float res = acc[mi][ni][r] + bs;
        if (RELU) res = fmaxf(res, 0.f);
        if (GATE) res = (gate[mr] == esel) ? res : 0.f;
        size_t ci = (size_t)mr * N + n;
        if (ACC)      C[ci] += res;
        else if (NTB) __builtin_nontemporal_store(res, &C[ci]);
        else          C[ci] = res;
      }
    }
}

// ---------------------------------------------------------------------------
// Causal attention, online softmax. One thread = one query row (HD=64 in
// registers); block of 128 queries stages 64-key K/V tiles through LDS.
// ---------------------------------------------------------------------------
__global__ __launch_bounds__(128)
void k_attn(const float* __restrict__ q, const float* __restrict__ k,
            const float* __restrict__ v, float* __restrict__ o) {
  __shared__ float Ks[64][64];
  __shared__ float Vs[64][64];
  const int nqb = TT / 128;                    // 8
  int bid = blockIdx.x;
  int qb = bid % nqb;
  int h  = (bid / nqb) % HH;
  int b  = bid / (nqb * HH);
  int t0 = qb * 128;
  int t  = t0 + threadIdx.x;

  const float* qrow = q + ((size_t)(b * TT + t) * CC) + h * HD;
  float qreg[HD], accv[HD];
#pragma unroll
  for (int d = 0; d < HD; d += 4) {
    float4 qv = *(const float4*)&qrow[d];
    qreg[d] = qv.x; qreg[d+1] = qv.y; qreg[d+2] = qv.z; qreg[d+3] = qv.w;
    accv[d] = 0.f; accv[d+1] = 0.f; accv[d+2] = 0.f; accv[d+3] = 0.f;
  }
  float mcur = -INFINITY, ssum = 0.f;

  for (int kt0 = 0; kt0 < t0 + 128; kt0 += 64) {
#pragma unroll
    for (int i = 0; i < 8; ++i) {              // 1024 float4 quads / 128 thr
      int e = i * 128 + threadIdx.x;
      int j = e >> 4, d4 = (e & 15) * 4;
      size_t src = ((size_t)(b * TT + kt0 + j) * CC) + h * HD + d4;
      *(float4*)&Ks[j][d4] = *(const float4*)&k[src];
      *(float4*)&Vs[j][d4] = *(const float4*)&v[src];
    }
    __syncthreads();
    int jmax = t - kt0; if (jmax > 63) jmax = 63;
    for (int j = 0; j <= jmax; ++j) {
      float s = 0.f;
#pragma unroll
      for (int d = 0; d < HD; ++d) s += qreg[d] * Ks[j][d];
      s *= 0.125f;                               // 1/sqrt(64)
      float mn = fmaxf(mcur, s);
      float corr = __expf(mcur - mn);
      float p = __expf(s - mn);
      ssum = ssum * corr + p;
#pragma unroll
      for (int d = 0; d < HD; ++d) accv[d] = accv[d] * corr + p * Vs[j][d];
      mcur = mn;
    }
    __syncthreads();
  }
  float* orow = o + ((size_t)(b * TT + t) * CC) + h * HD;
  float inv = 1.f / ssum;
#pragma unroll
  for (int d = 0; d < HD; d += 4) {
    float4 r; r.x = accv[d] * inv; r.y = accv[d+1] * inv;
    r.z = accv[d+2] * inv; r.w = accv[d+3] * inv;
    *(float4*)&orow[d] = r;
  }
}

// ---------------------------------------------------------------------------
// Router: K=1 top-1 => gate weight is exactly 1 at argmax logit.
// One wave (32 lanes) per token; shfl_xor reduce over C.
// ---------------------------------------------------------------------------
__global__ void k_router(const float* __restrict__ xn, const float* __restrict__ Wr,
                         int* __restrict__ eid) {
  int n = blockIdx.x;
  int lane = threadIdx.x;     // blockDim = 32 (wave32)
  float p0 = 0.f, p1 = 0.f, p2 = 0.f, p3 = 0.f;
  for (int c = lane; c < CC; c += 32) {
    float xv = xn[(size_t)n * CC + c];
    const float4 w4 = *(const float4*)&Wr[(size_t)c * EE];   // E=4 contiguous
    p0 += xv * w4.x; p1 += xv * w4.y; p2 += xv * w4.z; p3 += xv * w4.w;
  }
  for (int off = 16; off > 0; off >>= 1) {
    p0 += __shfl_xor(p0, off, 32);
    p1 += __shfl_xor(p1, off, 32);
    p2 += __shfl_xor(p2, off, 32);
    p3 += __shfl_xor(p3, off, 32);
  }
  if (lane == 0) {
    int best = 0; float bv = p0;
    if (p1 > bv) { bv = p1; best = 1; }
    if (p2 > bv) { bv = p2; best = 2; }
    if (p3 > bv) { bv = p3; best = 3; }
    eid[n] = best;
  }
}

// ---------------------------------------------------------------------------
// Host-side launch. Workspace (fp32): x, xn, q, k, v, attn (each NTOK*CC),
// H (NTOK*FF), expert ids (NTOK ints)  => ~63 MB.
// ---------------------------------------------------------------------------
extern "C" void kernel_launch(void* const* d_in, const int* in_sizes, int n_in,
                              void* d_out, int out_size, void* d_ws, size_t ws_size,
                              hipStream_t stream) {
  (void)in_sizes; (void)n_in; (void)out_size; (void)ws_size;
  const int*   idx  = (const int*)  d_in[0];
  const float* tok  = (const float*)d_in[1];
  const float* pos  = (const float*)d_in[2];
  const float* ln1s = (const float*)d_in[3];
  const float* ln1b = (const float*)d_in[4];
  const float* Wq   = (const float*)d_in[5];
  const float* Wk   = (const float*)d_in[6];
  const float* Wv   = (const float*)d_in[7];
  const float* Wo   = (const float*)d_in[8];
  const float* bo   = (const float*)d_in[9];
  const float* ln2s = (const float*)d_in[10];
  const float* ln2b = (const float*)d_in[11];
  const float* Wr   = (const float*)d_in[12];
  const float* W1   = (const float*)d_in[13];
  const float* b1   = (const float*)d_in[14];
  const float* W2   = (const float*)d_in[15];
  const float* b2   = (const float*)d_in[16];
  const float* lnfs = (const float*)d_in[17];
  const float* lnfb = (const float*)d_in[18];
  const float* Wh   = (const float*)d_in[19];
  const float* bh   = (const float*)d_in[20];
  float* out = (float*)d_out;

  float* ws = (float*)d_ws;
  float* x  = ws; ws += (size_t)NTOK * CC;
  float* xn = ws; ws += (size_t)NTOK * CC;
  float* qb = ws; ws += (size_t)NTOK * CC;
  float* kb = ws; ws += (size_t)NTOK * CC;
  float* vb = ws; ws += (size_t)NTOK * CC;
  float* ab = ws; ws += (size_t)NTOK * CC;
  float* Hb = ws; ws += (size_t)NTOK * FF;
  int*  eid = (int*)ws;

  const dim3 gCC(CC / BN, NTOK / BM);     // 12 x 32
  const dim3 gFF(FF / BN, NTOK / BM);     // 48 x 32
  const dim3 gVV(VV / BN, NTOK / BM);     // 500 x 32

  k_embed<<<(NTOK * CC / 4 + 255) / 256, 256, 0, stream>>>(idx, tok, pos, x);

  for (int l = 0; l < LL; ++l) {
    // --- attention block ---
    k_ln<<<NTOK, 256, 0, stream>>>(x, ln1s + l * CC, ln1b + l * CC, xn);
    //                N,  K,  BIAS, RELU, ACC,  GATE, NTB
    k_gemm<CC, CC, false, false, false, false, false>
        <<<gCC, 128, 0, stream>>>(xn, Wq + (size_t)l * CC * CC, nullptr, qb, nullptr, 0);
    k_gemm<CC, CC, false, false, false, false, false>
        <<<gCC, 128, 0, stream>>>(xn, Wk + (size_t)l * CC * CC, nullptr, kb, nullptr, 0);
    k_gemm<CC, CC, false, false, false, false, false>
        <<<gCC, 128, 0, stream>>>(xn, Wv + (size_t)l * CC * CC, nullptr, vb, nullptr, 0);
    k_attn<<<BB * HH * (TT / 128), 128, 0, stream>>>(qb, kb, vb, ab);
    // x += attn @ Wo + bo   (residual via accumulate epilogue)
    k_gemm<CC, CC, true, false, true, false, false>
        <<<gCC, 128, 0, stream>>>(ab, Wo + (size_t)l * CC * CC, bo + l * CC, x, nullptr, 0);

    // --- MoE block (top-1 of 4; gate weight == 1 at argmax) ---
    k_ln<<<NTOK, 256, 0, stream>>>(x, ln2s + l * CC, ln2b + l * CC, xn);
    k_router<<<NTOK, 32, 0, stream>>>(xn, Wr + (size_t)l * CC * EE, eid);
    for (int e = 0; e < EE; ++e) {
      const float* W1e = W1 + ((size_t)(l * EE + e)) * CC * FF;
      const float* b1e = b1 + ((size_t)(l * EE + e)) * FF;
      const float* W2e = W2 + ((size_t)(l * EE + e)) * FF * CC;
      const float* b2e = b2 + ((size_t)(l * EE + e)) * CC;
      // H = relu(xn @ W1e + b1e)
      k_gemm<FF, CC, true, true, false, false, false>
          <<<gFF, 128, 0, stream>>>(xn, W1e, b1e, Hb, nullptr, 0);
      // x += gate_e(m) * (H @ W2e + b2e)
      k_gemm<CC, FF, true, false, true, true, false>
          <<<gCC, 128, 0, stream>>>(Hb, W2e, b2e, x, eid, e);
    }
  }

  k_ln<<<NTOK, 256, 0, stream>>>(x, lnfs, lnfb, xn);
  // logits: 2048 x 32000, streaming weights + output (bypass L2 via NT hints)
  k_gemm<VV, CC, true, false, false, false, true>
      <<<gVV, 128, 0, stream>>>(xn, Wh, bh, out, nullptr, 0);
}
